// TwoLevelV3HieSupCon_40999757807786
// MI455X (gfx1250) — matrix operand order repaired and checked by reading, not compiled, not run on hardware
//
#include <hip/hip_runtime.h>
#include <hip/hip_bf16.h>

typedef __attribute__((ext_vector_type(16))) __bf16 v16bf;
typedef __attribute__((ext_vector_type(8)))  float  v8f;
typedef unsigned short ushort_t;

#define N_TOT    8192
#define FD       128
#define NEGBIG   -1e30f
#define LN2F     0.69314718055994530942f
// features pre-scaled by sqrt((1/0.07) * log2(e)) so WMMA emits logits*log2(e)
#define CVT_SCALE 4.53983442f

__device__ __forceinline__ v16bf load_frag(const ushort_t* p0, const ushort_t* p1) {
    union { v16bf v; uint4 q[2]; } u;
    u.q[0] = *(const uint4*)p0;
    u.q[1] = *(const uint4*)p1;
    return u.v;
}

// Load the 16x128 A-tile fragments for this block's rows (K = 4 chunks of 32).
__device__ __forceinline__ void load_a_frags(const ushort_t* fb, int m0, int lrow,
                                             int half, v16bf afrag[4]) {
    const ushort_t* rowp = fb + (size_t)(m0 + lrow) * FD;
    #pragma unroll
    for (int c = 0; c < 4; ++c) {
        int kA = 32 * c + half * 8;
        afrag[c] = load_frag(rowp + kA, rowp + kA + 16);
    }
}

// One 16x16 logit tile (scaled by log2e): issue all 8 b128 loads first, then the
// 4-deep WMMA chain so loads overlap earlier WMMAs (staggered s_wait_loadcnt).
__device__ __forceinline__ v8f tile_wmma(const ushort_t* fb, const v16bf afrag[4],
                                         int j0, int lrow, int half) {
    const ushort_t* colp = fb + (size_t)(j0 + lrow) * FD;
    v16bf bfrag[4];
    #pragma unroll
    for (int ch = 0; ch < 4; ++ch) {
        int kB = 32 * ch + half * 8;
        bfrag[ch] = load_frag(colp + kB, colp + kB + 16);
    }
    v8f c = {};
    #pragma unroll
    for (int ch = 0; ch < 4; ++ch) {
        c = __builtin_amdgcn_wmma_f32_16x16x32_bf16(
                false, afrag[ch], false, bfrag[ch], (short)0, c, false, false);
    }
    return c;
}

// ---------------- Kernel 1: f32 -> scaled bf16 (RNE) + zero accumulators --------
__global__ __launch_bounds__(256) void cvt_bf16_kernel(const float* __restrict__ f,
                                                       ushort_t* __restrict__ fb,
                                                       float* __restrict__ acc) {
    int i = blockIdx.x * 256 + threadIdx.x;
    if (i < N_TOT * FD) {
        unsigned u = __float_as_uint(f[i] * CVT_SCALE);
        unsigned r = (u + 0x7FFFu + ((u >> 16) & 1u)) >> 16;   // RNE to bf16
        fb[i] = (ushort_t)r;
    }
    if (i < 2) acc[i] = 0.0f;
}

// ------- Kernel 2: fused two-sweep (row max, then sumexp/psum/nsum) ------------
__global__ __launch_bounds__(256) void supcon_kernel(const ushort_t* __restrict__ fb,
                                                     const int* __restrict__ labels,
                                                     float* __restrict__ acc) {
    __shared__ int   s_labels[512];
    __shared__ float s_red[8][16][3];   // [wave][row][(max) | (sumexp,psum,nsum)]
    __shared__ float s_rowmax[16];
    __shared__ int   s_cnt;

    const int tid  = threadIdx.x;
    // readfirstlane: wave index is wave-uniform -> SGPR, so per-tile branches and
    // loop control compile to scalar s_cmp/s_cbranch instead of EXEC masking.
    const int wave = __builtin_amdgcn_readfirstlane(tid >> 5);
    const int lane = tid & 31;
    const int lrow = lane & 15;
    const int half = lane >> 4;
    const int bx   = blockIdx.x;
    const int m0   = bx * 16;

    for (int i = tid; i < 512; i += 256) s_labels[i] = labels[i];
    if (tid == 0) s_cnt = 0;
    __syncthreads();

    const int lab_block = s_labels[bx];         // one batch -> one label per block
    {   // count of batches sharing this block's label
        int mycnt = 0;
        for (int i = tid; i < 512; i += 256) mycnt += (s_labels[i] == lab_block);
        atomicAdd(&s_cnt, mycnt);
    }

    v16bf afrag[4];
    load_a_frags(fb, m0, lrow, half, afrag);

    // ---------------- Sweep 1: per-row max over valid (different-batch) cols ----
    float rmax[8];
    #pragma unroll
    for (int v = 0; v < 8; ++v) rmax[v] = NEGBIG;

    for (int jt = wave; jt < 512; jt += 8) {
        if (jt == bx) continue;                 // diagonal tile: same batch, masked
        v8f c = tile_wmma(fb, afrag, jt * 16, lrow, half);
        #pragma unroll
        for (int v = 0; v < 8; ++v) rmax[v] = fmaxf(rmax[v], c[v]);
    }

    #pragma unroll
    for (int v = 0; v < 8; ++v) {
        float m = rmax[v];
        #pragma unroll
        for (int k = 1; k < 16; k <<= 1) m = fmaxf(m, __shfl_xor(m, k, 32));
        if (lrow == 0) s_red[wave][half * 8 + v][0] = m;
    }
    __syncthreads();

    if (wave == 0 && lane < 16) {
        float m = NEGBIG;
        #pragma unroll
        for (int w = 0; w < 8; ++w) m = fmaxf(m, s_red[w][lane][0]);
        s_rowmax[lane] = m;                     // broadcast via LDS
    }
    __syncthreads();

    const float pcnt = 16.0f * ((float)s_cnt - 1.0f);
    float M[8], ssum[8], psum[8], nsum[8];
    #pragma unroll
    for (int v = 0; v < 8; ++v) {
        M[v] = s_rowmax[half * 8 + v];          // log2e-scaled row max
        ssum[v] = 0.f; psum[v] = 0.f; nsum[v] = 0.f;
    }
    __syncthreads();   // s_red reused below

    // ---------------- Sweep 2: sumexp + class-positive sums + noise sums -------
    for (int jt = wave; jt < 512; jt += 8) {
        v8f c = tile_wmma(fb, afrag, jt * 16, lrow, half);

        if (jt == bx) {
            // Diagonal tile: only noise-level positives (same group, not self).
            // Static pattern: group match <=> (lrow>>3)==half; self <=> (lrow&7)==v.
            #pragma unroll
            for (int v = 0; v < 8; ++v) {
                bool posn = ((lrow >> 3) == half) && ((lrow & 7) != v);
                nsum[v] += posn ? c[v] : 0.0f;
            }
        } else {
            // All 256 elements valid; class-positive flag is tile-uniform.
            const float flag = (s_labels[jt] == lab_block) ? 1.0f : 0.0f;
            #pragma unroll
            for (int v = 0; v < 8; ++v) {
                ssum[v] += __builtin_amdgcn_exp2f(c[v] - M[v]);   // native v_exp_f32
                psum[v]  = fmaf(flag, c[v], psum[v]);
            }
        }
    }

    // Plain-sum lane reduction (max is fixed, no rescale needed).
    #pragma unroll
    for (int v = 0; v < 8; ++v) {
        float s = ssum[v], ps = psum[v], ns = nsum[v];
        #pragma unroll
        for (int k = 1; k < 16; k <<= 1) {
            s  += __shfl_xor(s,  k, 32);
            ps += __shfl_xor(ps, k, 32);
            ns += __shfl_xor(ns, k, 32);
        }
        if (lrow == 0) {
            float* dst = s_red[wave][half * 8 + v];
            dst[0] = s; dst[1] = ps; dst[2] = ns;
        }
    }
    __syncthreads();

    if (wave == 0 && lane < 16) {
        float s = 0.f, ps = 0.f, ns = 0.f;
        #pragma unroll
        for (int w = 0; w < 8; ++w) {
            const float* src = s_red[w][lane];
            s += src[0]; ps += src[1]; ns += src[2];
        }
        // Convert from log2e-scaled logits back to natural units.
        const float lse = (s_rowmax[lane] + __builtin_amdgcn_logf(s)) * LN2F;
        float cterm = (pcnt * lse - ps * LN2F) / (pcnt + 1e-8f);
        float nrow  = ns * LN2F;
        #pragma unroll
        for (int k = 1; k < 16; k <<= 1) {
            cterm += __shfl_xor(cterm, k, 32);
            nrow  += __shfl_xor(nrow,  k, 32);
        }
        if (lane == 0) {
            atomicAdd(&acc[0], cterm);
            atomicAdd(&acc[1], nrow);
        }
    }
}

// ---------------- Kernel 3: finalize scalar loss ----------------
__global__ void finalize_kernel(const float* __restrict__ acc, float* __restrict__ out) {
    if (threadIdx.x == 0 && blockIdx.x == 0) {
        const float loss_class = acc[0] / (float)N_TOT;
        const float loss_noise = -(acc[1] / ((float)N_TOT * 7.0f)) / 0.07f;
        out[0] = (loss_class + loss_noise) * (1.0f / 3.0f);  // alpha=1/3, aug=0, T/Tb=1
    }
}

extern "C" void kernel_launch(void* const* d_in, const int* in_sizes, int n_in,
                              void* d_out, int out_size, void* d_ws, size_t ws_size,
                              hipStream_t stream) {
    const float* features = (const float*)d_in[0];   // 8192x128 f32
    const int*   labels   = (const int*)d_in[1];     // 512 i32

    ushort_t* fb  = (ushort_t*)d_ws;                                  // 2 MB bf16
    float*    acc = (float*)((char*)d_ws + (size_t)N_TOT * FD * 2);   // 2 floats

    const int n_elems = N_TOT * FD;
    cvt_bf16_kernel<<<(n_elems + 255) / 256, 256, 0, stream>>>(features, fb, acc);
    supcon_kernel<<<N_TOT / 16, 256, 0, stream>>>(fb, labels, acc);
    finalize_kernel<<<1, 32, 0, stream>>>(acc, (float*)d_out);
}
